// TriangleSelfAttentionRowWise_16458314678291
// MI455X (gfx1250) — matrix operand order repaired and checked by reading, not compiled, not run on hardware
//
#include <hip/hip_runtime.h>

// ---------------------------------------------------------------------------
// Types for CDNA5 WMMA (wave32)
// ---------------------------------------------------------------------------
typedef __attribute__((ext_vector_type(16))) __bf16 v16bf;
typedef __attribute__((ext_vector_type(8)))  __bf16 v8bf;
typedef __attribute__((ext_vector_type(4)))  __bf16 v4bf;
typedef __attribute__((ext_vector_type(2)))  __bf16 v2bf;
typedef __attribute__((ext_vector_type(8)))  float  v8f;

static constexpr int NN  = 384;  // N
static constexpr int CC  = 128;  // C
static constexpr int RB  = 3;    // row-blocks (of 16) per workgroup
static constexpr int R   = RB * 16;   // rows per workgroup = 48
static constexpr int P1  = 132;  // pitch (floats)  for 128-wide f32 LDS buffers
static constexpr int PZB = 136;  // pitch (ushorts) for 128-wide bf16 LDS (mult of 8)
static constexpr int POB = 392;  // pitch (ushorts) for 384-wide bf16 LDS (mult of 8)

#if defined(__has_builtin)
#if __has_builtin(__builtin_amdgcn_cvt_pk_bf16_f32)
#define HAVE_CVT_PK_BF16 1
#endif
#endif

__device__ __forceinline__ __bf16 f2bf(float f) {
  union { float f; unsigned u; } x; x.f = f;
  unsigned r = x.u + 0x7FFFu + ((x.u >> 16) & 1u);   // round-to-nearest-even
  return __builtin_bit_cast(__bf16, (unsigned short)(r >> 16));
}
// Pack two f32 -> two bf16 (single v_cvt_pk_bf16_f32 when available).
__device__ __forceinline__ v2bf f2bf2(float a, float b) {
#ifdef HAVE_CVT_PK_BF16
  auto t = __builtin_amdgcn_cvt_pk_bf16_f32(a, b);
  return __builtin_bit_cast(v2bf, t);
#else
  v2bf r; r[0] = f2bf(a); r[1] = f2bf(b); return r;
#endif
}
__device__ __forceinline__ unsigned short f2bfu(float f) {
  return __builtin_bit_cast(unsigned short, f2bf2(f, f)[0]);
}
__device__ __forceinline__ float bf2f(unsigned short us) {
  union { unsigned u; float f; } x; x.u = ((unsigned)us) << 16;
  return x.f;
}

// A-operand fragment (16x32 bf16) from a bf16 LDS tile: two contiguous 16B loads.
// ISA 7.12.2 A layout: lane L row M=L%16; VGPRs 0-3 hold K = 8*(L/16)+0..7,
// VGPRs 4-7 hold K = 8*(L/16)+16..23.
__device__ __forceinline__ v16bf load_a(const unsigned short* A, int pitch,
                                        int rowOff, int kbase, int lane) {
  const unsigned short* p = A + (rowOff + (lane & 15)) * pitch + kbase + ((lane >> 4) << 3);
  union { struct { v8bf lo, hi; } s; v16bf v; } u;
  u.s.lo = *reinterpret_cast<const v8bf*>(p);
  u.s.hi = *reinterpret_cast<const v8bf*>(p + 16);
  return u.v;
}

// B-operand fragment from the pre-swizzled bf16 weight blob (32B per lane).
__device__ __forceinline__ v16bf ldB(const unsigned short* Wb, int nb, int kb, int lane) {
  return *reinterpret_cast<const v16bf*>(Wb + (((size_t)(nb * 4 + kb)) << 9) + (lane << 4));
}

#define WMMA(a, b, c) __builtin_amdgcn_wmma_f32_16x16x32_bf16(false, (a), false, (b), (short)0, (c), false, false)

// ---------------------------------------------------------------------------
// Prep: f32 weight [128 x Ncols] -> bf16 WMMA-B fragment order.
// B layout: lane L = column n=nb*16+L%16; K = kb*32 + 16*(e/8) + 8*(L/16) + e%8 (e=0..15).
// mode 1 builds the block-diagonal expansion of Wdd [32x96] -> [128x384].
// ---------------------------------------------------------------------------
__global__ __launch_bounds__(256) void swizzle_w(const float* __restrict__ W,
                                                 unsigned short* __restrict__ dst,
                                                 int Ncols, int mode) {
  const int t = blockIdx.x * 256 + threadIdx.x;
  if (t >= 128 * Ncols) return;
  const int frag = t >> 9, within = t & 511;
  const int lane = within >> 4, e = within & 15;
  const int nb = frag >> 2, kb = frag & 3;
  const int k = kb * 32 + ((e >> 3) << 4) + ((lane >> 4) << 3) + (e & 7);
  const int n = nb * 16 + (lane & 15);
  float val;
  if (mode == 0) {
    val = W[k * Ncols + n];
  } else {
    const int hk = k >> 5, hn = n / 96;
    val = (hk == hn) ? W[(k & 31) * 96 + (n % 96)] : 0.0f;
  }
  dst[t] = f2bfu(val);
}

// ---------------------------------------------------------------------------
// vec path -> vec_dot [384 x 128] in workspace
// ---------------------------------------------------------------------------
__global__ __launch_bounds__(192) void vec_kernel(const float* __restrict__ vec,
                                                  const float* __restrict__ g,
                                                  const float* __restrict__ b,
                                                  const float* __restrict__ Wvp,
                                                  const float* __restrict__ Wmap,
                                                  float* __restrict__ vd_out) {
  __shared__ float vn[3][64];
  __shared__ float vd[64];
  __shared__ float mv[3][2];
  const int n = blockIdx.x, t = threadIdx.x;
  const int m = t >> 6, k = t & 63;
  const float x = vec[(n * 3 + m) * 64 + k];
  vn[m][k] = x;
  __syncthreads();
  if (t < 3) {
    float s = 0.f, ss = 0.f;
    for (int i = 0; i < 64; ++i) { float y = vn[t][i]; s += y; ss += y * y; }
    const float mean = s * (1.f / 64.f);
    mv[t][0] = mean; mv[t][1] = rsqrtf(ss * (1.f / 64.f) - mean * mean + 1e-5f);
  }
  __syncthreads();
  vn[m][k] = (x - mv[m][0]) * mv[m][1] * g[k] + b[k];
  __syncthreads();
  if (t < 64) {
    float acc = 0.f;
    for (int mm = 0; mm < 3; ++mm) {
      float a1 = 0.f, a2 = 0.f;
      for (int kk = 0; kk < 64; ++kk) {
        const float vv = vn[mm][kk];
        a1 += vv * Wvp[kk * 192 + t];
        a2 += vv * Wvp[kk * 192 + 64 + t];
      }
      acc += a1 * a2;
    }
    vd[t] = acc;
  }
  __syncthreads();
  if (t < 128) {
    float acc = 0.f;
    for (int kk = 0; kk < 64; ++kk) acc += vd[kk] * Wmap[kk * 128 + t];
    vd_out[n * 128 + t] = acc;
  }
}

// ---------------------------------------------------------------------------
// Fused main kernel: one WG = 48 rows (3 row-blocks) of the flat [N*N, C] op.
// ---------------------------------------------------------------------------
__global__ __launch_bounds__(256) void fused_kernel(
    const float* __restrict__ z, const float* __restrict__ zmask,
    const float* __restrict__ ln_g, const float* __restrict__ ln_b,
    const unsigned short* __restrict__ Wq, const unsigned short* __restrict__ Wk,
    const unsigned short* __restrict__ Wv, const unsigned short* __restrict__ Wdd,
    const unsigned short* __restrict__ Wo, const float* __restrict__ bo,
    const unsigned short* __restrict__ Wg1, const float* __restrict__ bg1,
    const unsigned short* __restrict__ Wl1, const float* __restrict__ bl1,
    const unsigned short* __restrict__ Wg2, const float* __restrict__ bg2,
    const unsigned short* __restrict__ Wl2, const float* __restrict__ bl2,
    const unsigned short* __restrict__ Wf, const float* __restrict__ bfv,
    const float* __restrict__ vec_dot, float* __restrict__ out) {
  __shared__ unsigned short s_znb[R * PZB];  // bf16: LN(z); then x; then final A
  __shared__ unsigned short s_ob [R * POB];  // bf16: o = x@Wo + bo
  __shared__ unsigned short s_ab [R * POB];  // bf16: v = (z@Wv)*(dd@WddBig)
  __shared__ float s_s[R * P1];              // f32: s=q*k*dd+mask; then sig(g1)*l1
  __shared__ float s_p[R * P1];              // f32: sig(g2)*l2

  const int wave = threadIdx.x >> 5, lane = threadIdx.x & 31;
  const int r0 = blockIdx.x * R;
  const int jbase = r0 % NN;                 // 384 % 48 == 0 -> j = jbase + rrow
  const int m16 = lane & 15;
  const int h8  = (lane >> 4) << 3;

  // -------- S0: load 48 rows of z, LayerNorm over C=128, store bf16 ---------
#pragma unroll
  for (int rr = 0; rr < R / 8; ++rr) {
    const int lr = wave * (R / 8) + rr;
    const float4 x = reinterpret_cast<const float4*>(z + (size_t)(r0 + lr) * CC)[lane];
    float s  = x.x + x.y + x.z + x.w;
    float ss = x.x * x.x + x.y * x.y + x.z * x.z + x.w * x.w;
#pragma unroll
    for (int o = 16; o > 0; o >>= 1) { s += __shfl_xor(s, o, 32); ss += __shfl_xor(ss, o, 32); }
    const float mean = s * (1.f / 128.f);
    const float inv  = rsqrtf(ss * (1.f / 128.f) - mean * mean + 1e-5f);
    const float4 g = reinterpret_cast<const float4*>(ln_g)[lane];
    const float4 b = reinterpret_cast<const float4*>(ln_b)[lane];
    union { struct { v2bf a, b; } s; v4bf v; } pk;
    pk.s.a = f2bf2((x.x - mean) * inv * g.x + b.x, (x.y - mean) * inv * g.y + b.y);
    pk.s.b = f2bf2((x.z - mean) * inv * g.z + b.z, (x.w - mean) * inv * g.w + b.w);
    *reinterpret_cast<v4bf*>(s_znb + lr * PZB + lane * 4) = pk.v;
  }
  __syncthreads();

  // -------- S1: fused q,k GEMMs -> s = q*k*dd + mask (in-register) ----------
  {
    const int nb = wave;
    v8f aq[RB], ak[RB];
#pragma unroll
    for (int rb = 0; rb < RB; ++rb) { aq[rb] = (v8f)0.f; ak[rb] = (v8f)0.f; }
#pragma unroll
    for (int kb = 0; kb < 4; ++kb) {
      const v16bf bq = ldB(Wq, nb, kb, lane);
      const v16bf bk = ldB(Wk, nb, kb, lane);
#pragma unroll
      for (int rb = 0; rb < RB; ++rb) {
        const v16bf a = load_a(s_znb, PZB, rb * 16, kb * 32, lane);
        aq[rb] = WMMA(a, bq, aq[rb]);
        ak[rb] = WMMA(a, bk, ak[rb]);
      }
    }
    const int col = nb * 16 + m16;
#pragma unroll
    for (int rb = 0; rb < RB; ++rb)
#pragma unroll
      for (int v = 0; v < 8; ++v) {
        const int row = rb * 16 + h8 + v;
        const float znv = bf2f(s_znb[row * PZB + col]);
        s_s[row * P1 + col] = aq[rb][v] * ak[rb][v] * znv + zmask[r0 + row];
      }
  }
  // -------- S2: fused v3 = z@Wv, ddm = z@WddBig -> s_ab = bf16(v3*ddm) ------
  for (int nb = wave; nb < 24; nb += 8) {
    v8f av[RB], ad[RB];
#pragma unroll
    for (int rb = 0; rb < RB; ++rb) { av[rb] = (v8f)0.f; ad[rb] = (v8f)0.f; }
#pragma unroll
    for (int kb = 0; kb < 4; ++kb) {
      const v16bf bv = ldB(Wv,  nb, kb, lane);
      const v16bf bd = ldB(Wdd, nb, kb, lane);
#pragma unroll
      for (int rb = 0; rb < RB; ++rb) {
        const v16bf a = load_a(s_znb, PZB, rb * 16, kb * 32, lane);
        av[rb] = WMMA(a, bv, av[rb]);
        ad[rb] = WMMA(a, bd, ad[rb]);
      }
    }
    const int col = nb * 16 + m16;
#pragma unroll
    for (int rb = 0; rb < RB; ++rb)
#pragma unroll
      for (int v = 0; v < 8; ++v)
        s_ab[(rb * 16 + h8 + v) * POB + col] = f2bfu(av[rb][v] * ad[rb][v]);
  }
  __syncthreads();

  // -------- S3: softmax over c=32 per (row, head); x = attn*v -> bf16 -------
  for (int p = wave; p < R * 4; p += 8) {
    const int rrow = p >> 2, hh = p & 3;
    const float sv = s_s[rrow * P1 + hh * 32 + lane];
    float mx = sv;
#pragma unroll
    for (int o = 16; o > 0; o >>= 1) mx = fmaxf(mx, __shfl_xor(mx, o, 32));
    const float ex = __expf(sv - mx);
    float sum = ex;
#pragma unroll
    for (int o = 16; o > 0; o >>= 1) sum += __shfl_xor(sum, o, 32);
    const float xv = (ex / sum) * bf2f(s_ab[rrow * POB + hh * 96 + lane]);
    s_znb[rrow * PZB + hh * 32 + lane] = f2bfu(xv);
  }
  __syncthreads();

  // -------- S4: o = x@Wo + bo -> bf16 s_ob ----------------------------------
  for (int nb = wave; nb < 24; nb += 8) {
    v8f ao[RB];
#pragma unroll
    for (int rb = 0; rb < RB; ++rb) ao[rb] = (v8f)0.f;
#pragma unroll
    for (int kb = 0; kb < 4; ++kb) {
      const v16bf b = ldB(Wo, nb, kb, lane);
#pragma unroll
      for (int rb = 0; rb < RB; ++rb) {
        const v16bf a = load_a(s_znb, PZB, rb * 16, kb * 32, lane);
        ao[rb] = WMMA(a, b, ao[rb]);
      }
    }
    const int col = nb * 16 + m16;
    const float bias = bo[col];
#pragma unroll
    for (int rb = 0; rb < RB; ++rb)
#pragma unroll
      for (int v = 0; v < 8; ++v)
        s_ob[(rb * 16 + h8 + v) * POB + col] = f2bfu(ao[rb][v] + bias);
  }
  __syncthreads();

  // -------- S5: fused gating pairs: sig(oX@Wg+bg)*(oX@Wl+bl) ----------------
#pragma unroll
  for (int pr = 0; pr < 2; ++pr) {
    const unsigned short* Wgp = pr ? Wg2 : Wg1;
    const unsigned short* Wlp = pr ? Wl2 : Wl1;
    const float* bgp = pr ? bg2 : bg1;
    const float* blp = pr ? bl2 : bl1;
    float* dst = pr ? s_p : s_s;               // s (logits) dead -> reuse s_s
    const int aoff = pr ? 256 : 128;
    v8f ag[RB], al[RB];
#pragma unroll
    for (int rb = 0; rb < RB; ++rb) { ag[rb] = (v8f)0.f; al[rb] = (v8f)0.f; }
#pragma unroll
    for (int kb = 0; kb < 4; ++kb) {
      const v16bf bg = ldB(Wgp, wave, kb, lane);
      const v16bf bl = ldB(Wlp, wave, kb, lane);
#pragma unroll
      for (int rb = 0; rb < RB; ++rb) {
        const v16bf a = load_a(s_ob, POB, rb * 16, aoff + kb * 32, lane);
        ag[rb] = WMMA(a, bg, ag[rb]);
        al[rb] = WMMA(a, bl, al[rb]);
      }
    }
    const int col = wave * 16 + m16;
    const float bgv = bgp[col], blv = blp[col];
#pragma unroll
    for (int rb = 0; rb < RB; ++rb)
#pragma unroll
      for (int v = 0; v < 8; ++v) {
        const float gg = 1.f / (1.f + __expf(-(ag[rb][v] + bgv)));
        dst[(rb * 16 + h8 + v) * P1 + col] = gg * (al[rb][v] + blv);
      }
  }
  __syncthreads();

  // -------- S6: A = o1 * vec_dot * (1 + p2*v1 + p3*v2) -> bf16 s_znb --------
  for (int e = threadIdx.x; e < R * 128; e += 256) {
    const int rrow = e >> 7, cc = e & 127;
    const int hh = cc >> 5, c32 = cc & 31;
    const float v1 = bf2f(s_ab[rrow * POB + hh * 96 + 32 + c32]);
    const float v2 = bf2f(s_ab[rrow * POB + hh * 96 + 64 + c32]);
    const float t2 = s_s[rrow * P1 + cc] * v1;
    const float t3 = s_p[rrow * P1 + cc] * v2;
    const float o1 = bf2f(s_ob[rrow * POB + cc]);
    const float vd = vec_dot[(jbase + rrow) * 128 + cc];
    s_znb[rrow * PZB + cc] = f2bfu(o1 * vd * (1.f + t2 + t3));
  }
  __syncthreads();

  // -------- S7: dx = A@Wf + bf ; out = dx * mask ----------------------------
  {
    v8f af[RB];
#pragma unroll
    for (int rb = 0; rb < RB; ++rb) af[rb] = (v8f)0.f;
#pragma unroll
    for (int kb = 0; kb < 4; ++kb) {
      const v16bf b = ldB(Wf, wave, kb, lane);
#pragma unroll
      for (int rb = 0; rb < RB; ++rb) {
        const v16bf a = load_a(s_znb, PZB, rb * 16, kb * 32, lane);
        af[rb] = WMMA(a, b, af[rb]);
      }
    }
    const int nn = wave * 16 + m16;
    const float bias = bfv[nn];
#pragma unroll
    for (int rb = 0; rb < RB; ++rb)
#pragma unroll
      for (int v = 0; v < 8; ++v) {
        const int gr = r0 + rb * 16 + h8 + v;
        out[(size_t)gr * CC + nn] = (af[rb][v] + bias) * zmask[gr];
      }
  }
}

// ---------------------------------------------------------------------------
extern "C" void kernel_launch(void* const* d_in, const int* in_sizes, int n_in,
                              void* d_out, int out_size, void* d_ws, size_t ws_size,
                              hipStream_t stream) {
  (void)in_sizes; (void)n_in; (void)out_size; (void)ws_size;
  const float* z     = (const float*)d_in[0];
  const float* zmask = (const float*)d_in[1];
  const float* vec   = (const float*)d_in[2];
  const float* ln_g  = (const float*)d_in[3];
  const float* ln_b  = (const float*)d_in[4];
  const float* ln1_g = (const float*)d_in[5];
  const float* ln1_b = (const float*)d_in[6];
  const float* Wq    = (const float*)d_in[7];
  const float* Wk    = (const float*)d_in[8];
  const float* Wv    = (const float*)d_in[9];
  const float* Wvp   = (const float*)d_in[10];
  const float* Wmap  = (const float*)d_in[11];
  const float* Wdd   = (const float*)d_in[12];
  const float* Wo    = (const float*)d_in[13];
  const float* bo    = (const float*)d_in[14];
  const float* Wg1   = (const float*)d_in[15];
  const float* bg1   = (const float*)d_in[16];
  const float* Wl1   = (const float*)d_in[17];
  const float* bl1   = (const float*)d_in[18];
  const float* Wg2   = (const float*)d_in[19];
  const float* bg2   = (const float*)d_in[20];
  const float* Wl2   = (const float*)d_in[21];
  const float* bl2   = (const float*)d_in[22];
  const float* Wf    = (const float*)d_in[23];
  const float* bf_   = (const float*)d_in[24];
  float* out = (float*)d_out;

  // Workspace: [vec_dot f32 384*128][bf16 weight blobs ~512 KB]
  float* vec_dot = (float*)d_ws;
  unsigned short* blob = (unsigned short*)((char*)d_ws + (size_t)NN * CC * 4);
  unsigned short* bWq  = blob;
  unsigned short* bWk  = blob + 16384;
  unsigned short* bWv  = blob + 32768;
  unsigned short* bWdd = blob + 81920;
  unsigned short* bWo  = blob + 131072;
  unsigned short* bWg1 = blob + 180224;
  unsigned short* bWl1 = blob + 196608;
  unsigned short* bWg2 = blob + 212992;
  unsigned short* bWl2 = blob + 229376;
  unsigned short* bWf  = blob + 245760;

  swizzle_w<<<64,  256, 0, stream>>>(Wq,  bWq,  128, 0);
  swizzle_w<<<64,  256, 0, stream>>>(Wk,  bWk,  128, 0);
  swizzle_w<<<192, 256, 0, stream>>>(Wv,  bWv,  384, 0);
  swizzle_w<<<192, 256, 0, stream>>>(Wdd, bWdd, 384, 1);
  swizzle_w<<<192, 256, 0, stream>>>(Wo,  bWo,  384, 0);
  swizzle_w<<<64,  256, 0, stream>>>(Wg1, bWg1, 128, 0);
  swizzle_w<<<64,  256, 0, stream>>>(Wl1, bWl1, 128, 0);
  swizzle_w<<<64,  256, 0, stream>>>(Wg2, bWg2, 128, 0);
  swizzle_w<<<64,  256, 0, stream>>>(Wl2, bWl2, 128, 0);
  swizzle_w<<<64,  256, 0, stream>>>(Wf,  bWf,  128, 0);

  vec_kernel<<<NN, 192, 0, stream>>>(vec, ln1_g, ln1_b, Wvp, Wmap, vec_dot);

  fused_kernel<<<(NN * NN) / R, 256, 0, stream>>>(
      z, zmask, ln_g, ln_b, bWq, bWk, bWv, bWdd, bWo, bo,
      bWg1, bg1, bWl1, bl1, bWg2, bg2, bWl2, bl2, bWf, bf_, vec_dot, out);
}
